// MyEdgeAttConv_72086731096483
// MI455X (gfx1250) — matrix operand order repaired and checked by reading, not compiled, not run on hardware
//
#include <hip/hip_runtime.h>
#include <hip/hip_bf16.h>

typedef __attribute__((ext_vector_type(16))) _Float16 v16h;
typedef __attribute__((ext_vector_type(8)))  float    v8f;

union F16Frag { v16h v; _Float16 h[16]; };

// ---------------------------------------------------------------------------
// Pack a 128x128 fp32 weight matrix W[k][n] (row-major, k = reduction dim)
// into WMMA B-fragment order (f16).
// B 32x16 f16 layout (per ISA 7.12.2 / sparse B analogy):
//   lane<16 : col = lane,    h[j] = K = 32*s + j (j<8), 32*s + 16 + (j-8) (j>=8)
//   lane>=16: col = lane-16, h[j] = K = 32*s + 8 + j (j<8), 32*s + 24 + (j-8) (j>=8)
// Fragment storage index: ((t*4 + s)*32 + lane)*16 + j   (t = n-tile 0..7, s = k-step 0..3)
// ---------------------------------------------------------------------------
__global__ void pack_w_kernel(const float* __restrict__ W, _Float16* __restrict__ P) {
    int tid = blockIdx.x * blockDim.x + threadIdx.x;     // 0..1023
    if (tid >= 1024) return;
    int lane = tid & 31;
    int s    = (tid >> 5) & 3;
    int t    = tid >> 7;
    int n     = 16 * t + (lane & 15);
    int kbase = 32 * s + (lane >> 4) * 8;
    _Float16* dst = P + ((size_t)(t * 4 + s) * 32 + lane) * 16;
#pragma unroll
    for (int j = 0; j < 16; ++j)
        dst[j] = (_Float16)W[(size_t)(kbase + j + (j >= 8 ? 8 : 0)) * 128 + n];
}

__global__ void zero_u32_kernel(unsigned* __restrict__ p, long n) {
    long i = (long)blockIdx.x * blockDim.x + threadIdx.x;
    long stride = (long)gridDim.x * blockDim.x;
    for (; i < n; i += stride) p[i] = 0u;
}

// ---------------------------------------------------------------------------
// Y[R,128] = X[R,128] @ W + bias, W pre-packed f16 fragments staged in LDS,
// fp32 accumulate. One wave per 16-row tile, 256 threads (8 waves) per block.
// A 16x32 f16 layout (ISA 7.12.2):
//   lane row = lane&15; off8 = (lane>>4)*8
//   h[j]   (j<8) = K = 32*s + off8 + j
//   h[8+j] (j<8) = K = 32*s + 16 + off8 + j
// ---------------------------------------------------------------------------
__global__ void gemm128_bias_kernel(const float* __restrict__ X,
                                    const _Float16* __restrict__ packW,
                                    const float* __restrict__ bias,
                                    float* __restrict__ Y, long R) {
    __shared__ __align__(32) _Float16 sW[128 * 128];     // 32 KB packed weights
    {
        const uint4* src = (const uint4*)packW;
        uint4*       dst = (uint4*)sW;
#pragma unroll
        for (int i = 0; i < 8; ++i)                      // 2048 uint4 / 256 threads
            dst[threadIdx.x + 256 * i] = src[threadIdx.x + 256 * i];
    }
    __syncthreads();

    int  lane = threadIdx.x & 31;
    long tile = (long)blockIdx.x * (blockDim.x >> 5) + (threadIdx.x >> 5);
    long base = tile * 16;
    if (base >= R) return;                       // wave-uniform exit (after barrier)
    long row = base + (lane & 15);
    if (row >= R) row = R - 1;                   // clamp (tail stores guarded)
    const float* xr = X + row * 128;
    int off8 = (lane >> 4) * 8;

    F16Frag A[4];
#pragma unroll
    for (int s = 0; s < 4; ++s) {
        int k0 = 32 * s + off8, k1 = 32 * s + 16 + off8;
#pragma unroll
        for (int j = 0; j < 8; ++j) {
            A[s].h[j]     = (_Float16)xr[k0 + j];
            A[s].h[8 + j] = (_Float16)xr[k1 + j];
        }
    }
    const v16h* bp = (const v16h*)sW;
    bool full = (base + 16 <= R);                // wave-uniform
#pragma unroll
    for (int t = 0; t < 8; ++t) {
        v8f acc = {};
#pragma unroll
        for (int s = 0; s < 4; ++s) {
            v16h b = bp[(t * 4 + s) * 32 + lane];
            acc = __builtin_amdgcn_wmma_f32_16x16x32_f16(false, A[s].v, false, b,
                                                         (short)0, acc, false, false);
        }
        int col = 16 * t + (lane & 15);
        float bv = bias[col];
        long rbase = base + ((lane >> 4) ? 8 : 0);
        if (full) {                              // hot path: no EXEC toggling
#pragma unroll
            for (int rr = 0; rr < 8; ++rr)
                Y[(rbase + rr) * 128 + col] = acc[rr] + bv;
        } else {
#pragma unroll
            for (int rr = 0; rr < 8; ++rr)
                if (rbase + rr < R) Y[(rbase + rr) * 128 + col] = acc[rr] + bv;
        }
    }
}

// ---------------------------------------------------------------------------
// Per-edge attention logit: a[m] = leaky_relu((x[rol]-x[col])@Wa1 + ba1) @ Wa2 + ba2
// One wave per 16-edge tile; Wa1 staged in LDS; h@Wa2 fused on the accumulators.
// ---------------------------------------------------------------------------
__global__ void edge_att_kernel(const float* __restrict__ x,
                                const int* __restrict__ rol,
                                const int* __restrict__ colA,
                                const _Float16* __restrict__ packA1,
                                const float* __restrict__ ba1,
                                const float* __restrict__ wa2,
                                const float* __restrict__ ba2,
                                float* __restrict__ aOut,
                                long E, long M) {
    __shared__ __align__(32) _Float16 sW[128 * 128];     // 32 KB packed Wa1
    {
        const uint4* src = (const uint4*)packA1;
        uint4*       dst = (uint4*)sW;
#pragma unroll
        for (int i = 0; i < 8; ++i)
            dst[threadIdx.x + 256 * i] = src[threadIdx.x + 256 * i];
    }
    __syncthreads();

    int  lane = threadIdx.x & 31;
    long tile = (long)blockIdx.x * (blockDim.x >> 5) + (threadIdx.x >> 5);
    long base = tile * 16;
    if (base >= M) return;
    long m = base + (lane & 15);
    if (m >= M) m = M - 1;
    int r, c;
    if (m < E) { r = rol[m]; c = colA[m]; }
    else       { int i = (int)(m - E); r = i; c = i; }   // self loop
    const float* xr = x + (long)r * 128;
    const float* xc = x + (long)c * 128;
    int off8 = (lane >> 4) * 8;

    F16Frag A[4];
#pragma unroll
    for (int s = 0; s < 4; ++s) {
        int k0 = 32 * s + off8, k1 = 32 * s + 16 + off8;
#pragma unroll
        for (int j = 0; j < 8; ++j) {
            A[s].h[j]     = (_Float16)(xr[k0 + j] - xc[k0 + j]);
            A[s].h[8 + j] = (_Float16)(xr[k1 + j] - xc[k1 + j]);
        }
    }

    float p[8];
#pragma unroll
    for (int rr = 0; rr < 8; ++rr) p[rr] = 0.f;

    const v16h* bp = (const v16h*)sW;
#pragma unroll
    for (int t = 0; t < 8; ++t) {
        v8f acc = {};
#pragma unroll
        for (int s = 0; s < 4; ++s) {
            v16h b = bp[(t * 4 + s) * 32 + lane];
            acc = __builtin_amdgcn_wmma_f32_16x16x32_f16(false, A[s].v, false, b,
                                                         (short)0, acc, false, false);
        }
        int col = 16 * t + (lane & 15);
        float bias = ba1[col];
        float w2   = wa2[col];
#pragma unroll
        for (int rr = 0; rr < 8; ++rr) {
            float hv = acc[rr] + bias;
            hv = hv > 0.f ? hv : 0.01f * hv;     // leaky_relu(0.01)
            p[rr] += hv * w2;                    // partial dot with Wa2 for this column
        }
    }
    // reduce partials across each 16-lane half (xor masks <=8 stay inside a half)
#pragma unroll
    for (int rr = 0; rr < 8; ++rr) {
        float v = p[rr];
        v += __shfl_xor(v, 1, 32);
        v += __shfl_xor(v, 2, 32);
        v += __shfl_xor(v, 4, 32);
        v += __shfl_xor(v, 8, 32);
        p[rr] = v;                               // lane-uniform per half: half0=row rr, half1=row rr+8
    }
    float b2 = ba2[0];
    int sel = lane & 7;
    float v = p[0];
    if (sel == 1) v = p[1];
    if (sel == 2) v = p[2];
    if (sel == 3) v = p[3];
    if (sel == 4) v = p[4];
    if (sel == 5) v = p[5];
    if (sel == 6) v = p[6];
    if (sel == 7) v = p[7];
    long row = base + sel + ((lane >> 4) ? 8 : 0);
    if ((lane & 15) < 8 && row < M) aOut[row] = v + b2;
}

// monotonic float<->uint mapping for atomic max
__device__ __forceinline__ unsigned f2key(float f) {
    unsigned u = __float_as_uint(f);
    return (u & 0x80000000u) ? ~u : (u | 0x80000000u);
}
__device__ __forceinline__ float key2f(unsigned k) {
    unsigned u = (k & 0x80000000u) ? (k & 0x7FFFFFFFu) : ~k;
    return __uint_as_float(u);
}

__global__ void seg_max_kernel(const float* __restrict__ a, const int* __restrict__ colA,
                               unsigned* __restrict__ amaxU, long E, long M) {
    long m = (long)blockIdx.x * blockDim.x + threadIdx.x;
    if (m >= M) return;
    int c = (m < E) ? colA[m] : (int)(m - E);
    atomicMax(&amaxU[c], f2key(a[m]));
}

__global__ void seg_exp_sum_kernel(float* __restrict__ a, const int* __restrict__ colA,
                                   const unsigned* __restrict__ amaxU,
                                   float* __restrict__ denom, long E, long M) {
    long m = (long)blockIdx.x * blockDim.x + threadIdx.x;
    if (m >= M) return;
    int c = (m < E) ? colA[m] : (int)(m - E);
    float ea = __expf(a[m] - key2f(amaxU[c]));
    a[m] = ea;                                   // overwrite logits with exp values
    atomicAdd(&denom[c], ea);
}

__global__ void scatter_msg_kernel(const float* __restrict__ emb, const float* __restrict__ ea,
                                   const float* __restrict__ denom,
                                   const int* __restrict__ rol, const int* __restrict__ colA,
                                   float* __restrict__ msg, long E, long M) {
    long g = (long)blockIdx.x * blockDim.x + threadIdx.x;
    long m = g >> 7;
    int  k = (int)(g & 127);
    if (m >= M) return;
    int r, c;
    if (m < E) { r = rol[m]; c = colA[m]; }
    else       { int i = (int)(m - E); r = i; c = i; }
    float w = ea[m] / (denom[c] + 1e-16f);
    atomicAdd(&msg[(long)c * 128 + k], emb[(long)r * 128 + k] * w);
}

extern "C" void kernel_launch(void* const* d_in, const int* in_sizes, int n_in,
                              void* d_out, int out_size, void* d_ws, size_t ws_size,
                              hipStream_t stream) {
    (void)n_in; (void)out_size; (void)ws_size;
    const float* x     = (const float*)d_in[0];
    const int*   eidx  = (const int*)d_in[1];
    const float* W_emb = (const float*)d_in[2];
    const float* b_emb = (const float*)d_in[3];
    const float* Wa1   = (const float*)d_in[4];
    const float* ba1   = (const float*)d_in[5];
    const float* Wa2   = (const float*)d_in[6];
    const float* ba2   = (const float*)d_in[7];
    const float* W_upd = (const float*)d_in[8];
    const float* b_upd = (const float*)d_in[9];

    long N = in_sizes[0] / 128;
    long E = in_sizes[1] / 2;
    long M = E + N;
    const int* rol  = eidx;
    const int* colA = eidx + E;

    // workspace carve-out
    char* w = (char*)d_ws;
    auto carve = [&](size_t bytes) { char* p = w; w += (bytes + 255) & ~(size_t)255; return p; };
    float*     emb     = (float*)carve((size_t)N * 128 * 4);
    float*     msg     = (float*)carve((size_t)N * 128 * 4);
    float*     a       = (float*)carve((size_t)M * 4);
    unsigned*  amaxU   = (unsigned*)carve((size_t)N * 4);
    float*     denom   = (float*)carve((size_t)N * 4);
    _Float16*  packEmb = (_Float16*)carve(128 * 128 * 2);
    _Float16*  packA1  = (_Float16*)carve(128 * 128 * 2);
    _Float16*  packUpd = (_Float16*)carve(128 * 128 * 2);

    pack_w_kernel<<<4, 256, 0, stream>>>(W_emb, packEmb);
    pack_w_kernel<<<4, 256, 0, stream>>>(Wa1,   packA1);
    pack_w_kernel<<<4, 256, 0, stream>>>(W_upd, packUpd);

    zero_u32_kernel<<<2048, 256, 0, stream>>>((unsigned*)msg, N * 128);
    zero_u32_kernel<<<256,  256, 0, stream>>>(amaxU, N);
    zero_u32_kernel<<<256,  256, 0, stream>>>((unsigned*)denom, N);

    long tilesN = (N + 15) / 16;
    gemm128_bias_kernel<<<(unsigned)((tilesN + 7) / 8), 256, 0, stream>>>(x, packEmb, b_emb, emb, N);

    long tilesM = (M + 15) / 16;
    edge_att_kernel<<<(unsigned)((tilesM + 7) / 8), 256, 0, stream>>>(
        x, rol, colA, packA1, ba1, Wa2, ba2, a, E, M);

    unsigned mBlocks = (unsigned)((M + 255) / 256);
    seg_max_kernel<<<mBlocks, 256, 0, stream>>>(a, colA, amaxU, E, M);
    seg_exp_sum_kernel<<<mBlocks, 256, 0, stream>>>(a, colA, amaxU, denom, E, M);

    unsigned sBlocks = (unsigned)((M * 128 + 255) / 256);
    scatter_msg_kernel<<<sBlocks, 256, 0, stream>>>(emb, a, denom, rol, colA, msg, E, M);

    gemm128_bias_kernel<<<(unsigned)((tilesN + 7) / 8), 256, 0, stream>>>(msg, packUpd, b_upd, (float*)d_out, N);
}
